// MultiHeadAttention_73263552135218
// MI455X (gfx1250) — matrix operand order repaired and checked by reading, not compiled, not run on hardware
//
#include <hip/hip_runtime.h>

// MI455X (gfx1250) MHA forward: bf16 WMMA everywhere, f32 accumulate.
// B=2, S=2048, D=1024, H=16, Dh=64.
// Round 2: async global->LDS staging (gfx1250 GLOBAL_LOAD_ASYNC_TO_LDS_B128)
// for pure-copy tiles + global_prefetch for next K-tiles.

#define DM 1024
#define SL 2048
#define NH 16
#define DH 64

typedef __attribute__((ext_vector_type(16))) __bf16        v16bf;
typedef __attribute__((ext_vector_type(8)))  float         v8f;
typedef __attribute__((ext_vector_type(4)))  unsigned int  u32x4;
typedef __attribute__((ext_vector_type(4)))  float         f32x4;

union Frag {
  u32x4 u[2];   // two 16-byte halves
  v16bf v;      // WMMA A/B operand (8 VGPRs)
};

__device__ __forceinline__ unsigned short f2bf(float f) {
  unsigned int u = __builtin_bit_cast(unsigned int, f);
  u += 0x7fffu + ((u >> 16) & 1u);   // round-to-nearest-even
  return (unsigned short)(u >> 16);
}
__device__ __forceinline__ unsigned int pack2bf(float a, float b) {
  return (unsigned int)f2bf(a) | ((unsigned int)f2bf(b) << 16);
}

// gfx1250 async copy: global memory -> LDS, 16 bytes per lane, ASYNCcnt-tracked.
// Generic pointers to __shared__ carry the LDS byte offset in their low 32 bits
// (shared-aperture encoding), which is exactly what VDST of the async op wants.
__device__ __forceinline__ void async_copy16(void* lds_dst, const void* gsrc) {
  unsigned lds_off = (unsigned)(unsigned long long)(uintptr_t)lds_dst;
  asm volatile("global_load_async_to_lds_b128 %0, %1, off"
               :: "v"(lds_off), "v"(gsrc)
               : "memory");
}
__device__ __forceinline__ void wait_async0() {
  asm volatile("s_wait_asynccnt 0x0" ::: "memory");
}
__device__ __forceinline__ void lds_fence() {   // compiler fence; HW DS is
  asm volatile("" ::: "memory");                // in-order within a wave
  __builtin_amdgcn_wave_barrier();
}

// ---------------------------------------------------------------------------
// Kernel 1: qkv = x @ Wqkv^T + b ; scatter to Q (scaled 1/8, bf16),
// K^T ([B,H,Dh,S], bf16), V ([B,H,S,Dh], bf16).
// Block tile 128(M) x 64(N), K-step 32, 8 waves (16 rows each).
// ---------------------------------------------------------------------------
__global__ __launch_bounds__(256) void qkv_gemm_kernel(
    const float* __restrict__ x, const float* __restrict__ Wqkv,
    const float* __restrict__ bqkv,
    unsigned short* __restrict__ Qb, unsigned short* __restrict__ Ktb,
    unsigned short* __restrict__ Vb) {
  __shared__ alignas(16) unsigned short Asm[128 * 32];   // [m][k] bf16
  __shared__ alignas(16) unsigned short Bsm[32 * 64];    // [k][n] bf16
  const int tid  = threadIdx.x;
  const int lane = tid & 31;
  const int wv   = tid >> 5;
  const int m0   = blockIdx.y * 128;
  const int n0   = blockIdx.x * 64;
  const int arow = lane & 15;           // A-operand: row within 16-row slab
  const int k0   = (lane >> 4) * 8;     // ISA 16-bit A layout K chunk select

  v8f acc[4];
  for (int nc = 0; nc < 4; ++nc)
    for (int e = 0; e < 8; ++e) acc[nc][e] = 0.0f;

  const int ar = tid >> 1, ah = tid & 1;   // A-tile fill: 2 threads / row
  const int bn = tid >> 2, bq = tid & 3;   // B-tile fill: 4 threads / W row
  const float* xrow = x + (size_t)(m0 + ar) * DM + ah * 16;
  const float* wrow = Wqkv + (size_t)(n0 + bn) * DM + bq * 8;

  for (int kb = 0; kb < DM; kb += 32) {
    if (kb + 32 < DM) {   // uniform: prefetch next K-tile into caches
      __builtin_prefetch(xrow + kb + 32, 0, 1);
      __builtin_prefetch(wrow + kb + 32, 0, 1);
    }
    // ---- A tile: x[m0..m0+127][kb..kb+31] f32 -> bf16 LDS (row-major) ----
    {
      const f32x4* src = (const f32x4*)(xrow + kb);
      unsigned int* dst = (unsigned int*)&Asm[ar * 32 + ah * 16];
#pragma unroll
      for (int i = 0; i < 4; ++i) {
        f32x4 f = src[i];
        dst[2 * i]     = pack2bf(f.x, f.y);
        dst[2 * i + 1] = pack2bf(f.z, f.w);
      }
    }
    // ---- B tile: Wqkv[n0+bn][kb+bq*8 ..+7] transposed -> Bsm[k][n] ----
    {
      const f32x4* src = (const f32x4*)(wrow + kb);
      f32x4 f0 = src[0], f1 = src[1];
      float fv[8] = {f0.x, f0.y, f0.z, f0.w, f1.x, f1.y, f1.z, f1.w};
#pragma unroll
      for (int i = 0; i < 8; ++i) Bsm[(bq * 8 + i) * 64 + bn] = f2bf(fv[i]);
    }
    __syncthreads();
    // ---- per-wave WMMA: A 16x32, 4 B tiles of 32x16 ----
    Frag a;
    a.u[0] = *(const u32x4*)&Asm[(wv * 16 + arow) * 32 + k0];
    a.u[1] = *(const u32x4*)&Asm[(wv * 16 + arow) * 32 + 16 + k0];
#pragma unroll
    for (int nc = 0; nc < 4; ++nc) {
      Frag b;
#pragma unroll
      for (int i = 0; i < 16; ++i)
        b.v[i] = __builtin_bit_cast(__bf16, Bsm[(k0 + (i & 7) + (i >> 3) * 16) * 64 + nc * 16 + arow]);
      acc[nc] = __builtin_amdgcn_wmma_f32_16x16x32_bf16(
          false, a.v, false, b.v, (short)0, acc[nc], false, false);
    }
    __syncthreads();
  }

  // ---- epilogue: bias + scatter into Q / K^T / V ----
  const int hi = lane >> 4, col = lane & 15;
#pragma unroll
  for (int nc = 0; nc < 4; ++nc) {
#pragma unroll
    for (int e = 0; e < 8; ++e) {
      const int m_abs = m0 + wv * 16 + e + 8 * hi;   // C layout: row = e + 8*hi
      const int n_abs = n0 + nc * 16 + col;
      const float val = acc[nc][e] + bqkv[n_abs];
      const int part = n_abs >> 10;          // 0=Q 1=K 2=V
      const int h    = (n_abs & 1023) >> 6;
      const int d    = n_abs & 63;
      const int bb   = m_abs >> 11;          // batch (S=2048)
      const int s    = m_abs & 2047;
      const size_t bh = (size_t)(bb * NH + h);
      if (part == 0)      Qb [(bh * SL + s) * DH + d] = f2bf(val * 0.125f);
      else if (part == 1) Ktb[(bh * DH + d) * SL + s] = f2bf(val);
      else                Vb [(bh * SL + s) * DH + d] = f2bf(val);
    }
  }
}

// ---------------------------------------------------------------------------
// Kernel 2: causal flash attention. One wave owns a 16-row Q tile; waves are
// fully independent (per-wave LDS slices, no block barriers -> causal loop
// trip counts may differ per wave). kv processed in blocks of 32.
// K/V tiles staged with gfx1250 async global->LDS DMA.
// ---------------------------------------------------------------------------
__global__ __launch_bounds__(128) void flash_attn_kernel(
    const unsigned short* __restrict__ Qb, const unsigned short* __restrict__ Ktb,
    const unsigned short* __restrict__ Vb, unsigned short* __restrict__ attn) {
  __shared__ alignas(16) unsigned short Ksm[4][DH * 32];   // [d][kv] per wave
  __shared__ alignas(16) unsigned short Vsm[4][32 * DH];   // [kv][d] per wave
  __shared__ alignas(16) unsigned short Psm[4][16 * 32];   // P staging per wave
  const int tid  = threadIdx.x;
  const int lane = tid & 31;
  const int wv   = tid >> 5;
  const int bh   = blockIdx.y;                 // b*16 + h
  const int q0   = blockIdx.x * 64 + wv * 16;
  const int row  = lane & 15;
  const int hi   = lane >> 4;
  const int k0   = hi * 8;
  const int col  = row;

  unsigned short* Ks = Ksm[wv];
  unsigned short* Vs = Vsm[wv];
  unsigned short* Ps = Psm[wv];

  const unsigned short* Qp  = Qb  + ((size_t)bh * SL + q0) * DH;
  const unsigned short* Ktp = Ktb + (size_t)bh * DH * SL;
  const unsigned short* Vp  = Vb  + (size_t)bh * SL * DH;

  // Q A-operands: two 16x32 fragments covering d=0..31 and d=32..63
  Frag aq[2];
#pragma unroll
  for (int s = 0; s < 2; ++s) {
    aq[s].u[0] = *(const u32x4*)&Qp[row * DH + s * 32 + k0];
    aq[s].u[1] = *(const u32x4*)&Qp[row * DH + s * 32 + 16 + k0];
  }

  v8f acc[4];
  float mrow[8], lrow[8];
#pragma unroll
  for (int e = 0; e < 8; ++e) {
    mrow[e] = -__builtin_inff();
    lrow[e] = 0.0f;
    for (int nc = 0; nc < 4; ++nc) acc[nc][e] = 0.0f;
  }

  const int kv_end = q0 + 16;   // causal: last needed kv index is q0+15
  for (int kv0 = 0; kv0 < kv_end; kv0 += 32) {
    // ---- async-DMA K^T (64x32) and V (32x64) tiles into this wave's LDS ----
#pragma unroll
    for (int i = 0; i < 2; ++i) {
      const int d = lane * 2 + i;
#pragma unroll
      for (int j = 0; j < 4; ++j)
        async_copy16(&Ks[d * 32 + j * 8], &Ktp[(size_t)d * SL + kv0 + j * 8]);
    }
#pragma unroll
    for (int j = 0; j < 8; ++j)
      async_copy16(&Vs[lane * DH + j * 8], &Vp[(size_t)(kv0 + lane) * DH + j * 8]);
    wait_async0();
    lds_fence();

    // ---- S = Q K^T : 2 n-chunks x 2 k-steps of WMMA ----
    v8f sc[2];
#pragma unroll
    for (int t = 0; t < 2; ++t)
#pragma unroll
      for (int e = 0; e < 8; ++e) sc[t][e] = 0.0f;
#pragma unroll
    for (int t = 0; t < 2; ++t) {
#pragma unroll
      for (int s = 0; s < 2; ++s) {
        Frag bk;                                 // B: column kv = lane%16, d in table order
#pragma unroll
        for (int i = 0; i < 16; ++i)
          bk.v[i] = __builtin_bit_cast(__bf16, Ks[(s * 32 + k0 + (i & 7) + (i >> 3) * 16) * 32 + t * 16 + col]);
        sc[t] = __builtin_amdgcn_wmma_f32_16x16x32_bf16(
            false, aq[s].v, false, bk.v, (short)0, sc[t], false, false);
      }
    }

    // ---- causal mask + online softmax (rows live in 16-lane halves) ----
#pragma unroll
    for (int e = 0; e < 8; ++e) {
      const int r_abs = q0 + e + 8 * hi;
#pragma unroll
      for (int t = 0; t < 2; ++t) {
        const int c_abs = kv0 + t * 16 + col;
        sc[t][e] = (c_abs <= r_abs) ? sc[t][e] : -__builtin_inff();
      }
      float tmax = fmaxf(sc[0][e], sc[1][e]);
#pragma unroll
      for (int m = 8; m >= 1; m >>= 1) tmax = fmaxf(tmax, __shfl_xor(tmax, m, 32));
      const float mnew = fmaxf(mrow[e], tmax);
      const float corr = __expf(mrow[e] - mnew);   // 0 on first block
      float rsum = 0.0f;
#pragma unroll
      for (int t = 0; t < 2; ++t) {
        const float p = __expf(sc[t][e] - mnew);   // masked -> exp(-inf) = 0
        sc[t][e] = p;
        rsum += p;
      }
#pragma unroll
      for (int m = 8; m >= 1; m >>= 1) rsum += __shfl_xor(rsum, m, 32);
      lrow[e] = lrow[e] * corr + rsum;
      mrow[e] = mnew;
#pragma unroll
      for (int nc = 0; nc < 4; ++nc) acc[nc][e] *= corr;
    }

    // ---- P: C-layout f32 -> A-layout bf16 via per-wave LDS staging ----
#pragma unroll
    for (int t = 0; t < 2; ++t)
#pragma unroll
      for (int e = 0; e < 8; ++e)
        Ps[(e + 8 * hi) * 32 + t * 16 + col] = f2bf(sc[t][e]);
    lds_fence();

    Frag ap;
    ap.u[0] = *(const u32x4*)&Ps[row * 32 + k0];
    ap.u[1] = *(const u32x4*)&Ps[row * 32 + 16 + k0];
#pragma unroll
    for (int nc = 0; nc < 4; ++nc) {
      Frag bv;
#pragma unroll
      for (int i = 0; i < 16; ++i)
        bv.v[i] = __builtin_bit_cast(__bf16, Vs[(k0 + (i & 7) + (i >> 3) * 16) * DH + nc * 16 + col]);
      acc[nc] = __builtin_amdgcn_wmma_f32_16x16x32_bf16(
          false, ap.v, false, bv.v, (short)0, acc[nc], false, false);
    }
    lds_fence();
  }

  // ---- normalize, write attn-out bf16 as [B, S, H*Dh] ----
  const int bb = bh >> 4, h = bh & 15;
#pragma unroll
  for (int nc = 0; nc < 4; ++nc) {
#pragma unroll
    for (int e = 0; e < 8; ++e) {
      const int s_abs = q0 + e + 8 * hi;
      const float val = acc[nc][e] / lrow[e];
      attn[((size_t)bb * SL + s_abs) * DM + h * DH + nc * 16 + col] = f2bf(val);
    }
  }
}

// ---------------------------------------------------------------------------
// Kernel 3: out = attn @ Wout^T + b_out  (f32 output).
// A tile is already bf16 -> async global->LDS DMA; B tile converts f32->bf16.
// ---------------------------------------------------------------------------
__global__ __launch_bounds__(256) void out_gemm_kernel(
    const unsigned short* __restrict__ attn, const float* __restrict__ Wout,
    const float* __restrict__ bout, float* __restrict__ out) {
  __shared__ alignas(16) unsigned short Asm[128 * 32];
  __shared__ alignas(16) unsigned short Bsm[32 * 64];
  const int tid  = threadIdx.x;
  const int lane = tid & 31;
  const int wv   = tid >> 5;
  const int m0   = blockIdx.y * 128;
  const int n0   = blockIdx.x * 64;
  const int arow = lane & 15;
  const int k0   = (lane >> 4) * 8;

  v8f acc[4];
  for (int nc = 0; nc < 4; ++nc)
    for (int e = 0; e < 8; ++e) acc[nc][e] = 0.0f;

  const int ar = tid >> 1, ah = tid & 1;
  const int bn = tid >> 2, bq = tid & 3;
  const unsigned short* arowp = attn + (size_t)(m0 + ar) * DM + ah * 16;
  const float* wrow = Wout + (size_t)(n0 + bn) * DM + bq * 8;

  for (int kb = 0; kb < DM; kb += 32) {
    if (kb + 32 < DM) {
      __builtin_prefetch(arowp + kb + 32, 0, 1);
      __builtin_prefetch(wrow + kb + 32, 0, 1);
    }
    // A tile (already bf16): async global->LDS, 2x16B per thread
    async_copy16(&Asm[ar * 32 + ah * 16],     arowp + kb);
    async_copy16(&Asm[ar * 32 + ah * 16 + 8], arowp + kb + 8);
    // B tile: Wout transposed + converted (needs VALU, stays VGPR-mediated)
    {
      const f32x4* src = (const f32x4*)(wrow + kb);
      f32x4 f0 = src[0], f1 = src[1];
      float fv[8] = {f0.x, f0.y, f0.z, f0.w, f1.x, f1.y, f1.z, f1.w};
#pragma unroll
      for (int i = 0; i < 8; ++i) Bsm[(bq * 8 + i) * 64 + bn] = f2bf(fv[i]);
    }
    wait_async0();
    __syncthreads();
    Frag a;
    a.u[0] = *(const u32x4*)&Asm[(wv * 16 + arow) * 32 + k0];
    a.u[1] = *(const u32x4*)&Asm[(wv * 16 + arow) * 32 + 16 + k0];
#pragma unroll
    for (int nc = 0; nc < 4; ++nc) {
      Frag b;
#pragma unroll
      for (int i = 0; i < 16; ++i)
        b.v[i] = __builtin_bit_cast(__bf16, Bsm[(k0 + (i & 7) + (i >> 3) * 16) * 64 + nc * 16 + arow]);
      acc[nc] = __builtin_amdgcn_wmma_f32_16x16x32_bf16(
          false, a.v, false, b.v, (short)0, acc[nc], false, false);
    }
    __syncthreads();
  }

  const int hi = lane >> 4, col = lane & 15;
#pragma unroll
  for (int nc = 0; nc < 4; ++nc) {
#pragma unroll
    for (int e = 0; e < 8; ++e) {
      const int m_abs = m0 + wv * 16 + e + 8 * hi;
      const int n_abs = n0 + nc * 16 + col;
      out[(size_t)m_abs * DM + n_abs] = acc[nc][e] + bout[n_abs];
    }
  }
}

// ---------------------------------------------------------------------------
extern "C" void kernel_launch(void* const* d_in, const int* in_sizes, int n_in,
                              void* d_out, int out_size, void* d_ws, size_t ws_size,
                              hipStream_t stream) {
  (void)in_sizes; (void)n_in; (void)out_size; (void)ws_size;
  const float* x    = (const float*)d_in[0];
  const float* Wqkv = (const float*)d_in[1];
  const float* bqkv = (const float*)d_in[2];
  const float* Wout = (const float*)d_in[3];
  const float* bout = (const float*)d_in[4];
  float* out = (float*)d_out;

  // Workspace (bf16 as ushort): Q, K^T, V, attn-out -> 4M elements each, 32 MB.
  unsigned short* Qb   = (unsigned short*)d_ws;
  unsigned short* Ktb  = Qb  + (size_t)4 * 1024 * 1024;
  unsigned short* Vb   = Ktb + (size_t)4 * 1024 * 1024;
  unsigned short* attn = Vb  + (size_t)4 * 1024 * 1024;

  // QKV projection: M=4096, N=3072 -> grid (48, 32), 8 waves/block
  qkv_gemm_kernel<<<dim3(48, 32), 256, 0, stream>>>(x, Wqkv, bqkv, Qb, Ktb, Vb);
  // Attention: 64 q-rows per block (4 waves x 16), per (b,h)
  flash_attn_kernel<<<dim3(SL / 64, 2 * NH), 128, 0, stream>>>(Qb, Ktb, Vb, attn);
  // Output projection: M=4096, N=1024 -> grid (16, 32)
  out_gemm_kernel<<<dim3(16, 32), 256, 0, stream>>>(attn, Wout, bout, out);
}
